// LinearAttention_91250875170835
// MI455X (gfx1250) — matrix operand order repaired and checked by reading, not compiled, not run on hardware
//
#include <hip/hip_runtime.h>
#include <hip/hip_bf16.h>

#if __has_include(<hip/amd_detail/amd_gfx1250_TDM.h>)
#define ATHENA_TDM_6ARG 1
#endif

// ---------------------------------------------------------------------------
// Types
// ---------------------------------------------------------------------------
typedef __attribute__((ext_vector_type(16))) __bf16 bf16x16;
typedef __attribute__((ext_vector_type(8)))  __bf16 bf16x8;
typedef __attribute__((ext_vector_type(8)))  float  f32x8;
typedef __attribute__((ext_vector_type(4)))  unsigned int u32x4;
typedef __attribute__((ext_vector_type(8)))  int i32x8;
typedef __attribute__((ext_vector_type(4)))  int i32x4;

__device__ __forceinline__ f32x8 zero8() {
    f32x8 z;
#pragma unroll
    for (int i = 0; i < 8; ++i) z[i] = 0.0f;
    return z;
}

__device__ __forceinline__ bf16x16 frag_cat(bf16x8 lo, bf16x8 hi) {
    bf16x16 r;
#pragma unroll
    for (int i = 0; i < 8; ++i) { r[i] = lo[i]; r[i + 8] = hi[i]; }
    return r;
}

// A-fragment (16x32 bf16): lane M = lane&15; element e -> K =
// (e<8?0:16) + 8*(lane>=16) + (e&7).  Two 16-byte LDS loads.
__device__ __forceinline__ bf16x16 ldsA(const __bf16* base, int pitch,
                                        int row0, int col0, int lane) {
    const __bf16* p = base + (row0 + (lane & 15)) * pitch + col0 + ((lane >> 4) << 3);
    return frag_cat(*(const bf16x8*)p, *(const bf16x8*)(p + 16));
}

// B-fragment (32x16 bf16), source stored as B^T row-major [n][k]:
// lane N = lane&15; element e -> K = (e<8?0:16) + 8*(lane>=16) + (e&7).
__device__ __forceinline__ bf16x16 ldsB(const __bf16* base, int pitch,
                                        int row0, int k0, int lane) {
    const __bf16* p = base + (row0 + (lane & 15)) * pitch + k0 + ((lane >> 4) << 3);
    return frag_cat(*(const bf16x8*)p, *(const bf16x8*)(p + 16));
}

// B-fragment built from fp32 LDS source (converted on the fly)
__device__ __forceinline__ bf16x16 ldsB_f32(const float* base, int pitch,
                                            int row0, int k0, int lane) {
    const float* p = base + (row0 + (lane & 15)) * pitch + k0 + ((lane >> 4) << 3);
    bf16x16 r;
#pragma unroll
    for (int i = 0; i < 16; ++i) r[i] = (__bf16)p[(i < 8) ? i : i + 8];
    return r;
}

__device__ __forceinline__ f32x8 wmma_bf16(bf16x16 a, bf16x16 b, f32x8 c) {
    return __builtin_amdgcn_wmma_f32_16x16x32_bf16(
        false, a, false, b, (short)0, c, false, false);
}

// phi(x) = elu(x) + 1
__device__ __forceinline__ float feature_map(float x) {
    return x > 0.0f ? x + 1.0f : __expf(x);
}

// ---------------------------------------------------------------------------
// TDM: async 2D bf16 tile load Global -> LDS with padding so a 64-elem row
// lands on a 72-elem LDS pitch (pad_interval code 4 = 32 dwords data,
// pad_amount code 3 = 4 dwords pad).
// ---------------------------------------------------------------------------
__device__ __forceinline__ void tdm_load_2d_bf16_padded(
    const void* g, unsigned ldsOff, int tileRows, int tileCols, int rowStrideElems)
{
    unsigned long long ga = (unsigned long long)(uintptr_t)g;
    u32x4 g0;
    g0[0] = 1u;                                            // count=1, user mode
    g0[1] = ldsOff;                                        // lds_addr (bytes)
    g0[2] = (unsigned)ga;                                  // global_addr lo
    g0[3] = ((unsigned)(ga >> 32) & 0x01ffffffu) | (2u << 30); // addr hi | type=2
    i32x8 g1;
    g1[0] = (1 << 16) | (1 << 20) | (4 << 22) | (3 << 25); // data_size=2B, pad on
    g1[1] = (tileCols & 0xffff) << 16;                     // tensor_dim0 lo16
    g1[2] = (((unsigned)tileCols >> 16) & 0xffff) | ((tileRows & 0xffff) << 16);
    g1[3] = (((unsigned)tileRows >> 16) & 0xffff) | ((tileCols & 0xffff) << 16); // tile_dim0
    g1[4] = (tileRows & 0xffff);                           // tile_dim1 (tile_dim2=0)
    g1[5] = rowStrideElems;                                // tensor_dim0_stride lo32
    g1[6] = 0;
    g1[7] = 0;
    i32x4 z4; z4[0] = 0; z4[1] = 0; z4[2] = 0; z4[3] = 0;
#ifdef ATHENA_TDM_6ARG
    i32x8 z8;
#pragma unroll
    for (int i = 0; i < 8; ++i) z8[i] = 0;
    __builtin_amdgcn_tensor_load_to_lds(g0, g1, z4, z4, z8, 0);
#else
    __builtin_amdgcn_tensor_load_to_lds(g0, g1, z4, z4, 0);
#endif
}

// ---------------------------------------------------------------------------
// Unified projection GEMM:  out[m][e] = act( sum_k A[m][k] * W[e][k] + b[e] )
// M = 2048, N = K = 1024.  Tile 128x128, 256 threads (8 waves),
// wave = 64x32 region = 4x2 tiles.  Double-buffered LDS, reg-prefetched.
//   aHeadGatherBf16 : A is bf16 workspace in (B*H, L, d) layout (out-proj)
//   outHeadScatterBf16 : result written bf16 to (B*H, L, d) workspace
// ---------------------------------------------------------------------------
#define GP 40   // LDS bf16 pitch for 32-wide K tiles (80 B rows, conflict-free)

__global__ __launch_bounds__(256) void proj_gemm_kernel(
    const void* __restrict__ Ain, const float* __restrict__ W,
    const float* __restrict__ bias, void* __restrict__ outp,
    int aHeadGatherBf16, int act, int outHeadScatterBf16)
{
    __shared__ alignas(16) __bf16 As[2][128][GP];
    __shared__ alignas(16) __bf16 Bs[2][128][GP];

    const int tid  = threadIdx.x;
    const int lane = tid & 31;
    const int w    = tid >> 5;     // 0..7
    const int wm   = w >> 2;       // 0..1 -> 64-row slab
    const int wn   = w & 3;        // 0..3 -> 32-col slab
    const int m0   = blockIdx.y * 128;
    const int n0   = blockIdx.x * 128;

    const int lr = tid >> 1;           // 0..127 : row within tile
    const int lc = (tid & 1) * 16;     // 0 or 16 : col group

    f32x8 acc[4][2];
#pragma unroll
    for (int i = 0; i < 4; ++i)
#pragma unroll
        for (int j = 0; j < 2; ++j) acc[i][j] = zero8();

    bf16x8 rA[2], rB[2];

    auto loadA = [&](int k0) {
        if (!aHeadGatherBf16) {
            const float* ap = (const float*)Ain + (size_t)(m0 + lr) * 1024 + k0 + lc;
#pragma unroll
            for (int h = 0; h < 2; ++h) {
                float4 v0 = *(const float4*)(ap + 8 * h);
                float4 v1 = *(const float4*)(ap + 8 * h + 4);
                rA[h][0] = (__bf16)v0.x; rA[h][1] = (__bf16)v0.y;
                rA[h][2] = (__bf16)v0.z; rA[h][3] = (__bf16)v0.w;
                rA[h][4] = (__bf16)v1.x; rA[h][5] = (__bf16)v1.y;
                rA[h][6] = (__bf16)v1.z; rA[h][7] = (__bf16)v1.w;
            }
        } else {
            const int m = m0 + lr, n = m & 1, l = m >> 1, k = k0 + lc;
            const __bf16* ap = (const __bf16*)Ain +
                ((size_t)((n * 16 + (k >> 6)) * 1024 + l)) * 64 + (k & 63);
            rA[0] = *(const bf16x8*)ap;
            rA[1] = *(const bf16x8*)(ap + 8);
        }
    };
    auto loadB = [&](int k0) {
        const float* bp = W + (size_t)(n0 + lr) * 1024 + k0 + lc;
#pragma unroll
        for (int h = 0; h < 2; ++h) {
            float4 v0 = *(const float4*)(bp + 8 * h);
            float4 v1 = *(const float4*)(bp + 8 * h + 4);
            rB[h][0] = (__bf16)v0.x; rB[h][1] = (__bf16)v0.y;
            rB[h][2] = (__bf16)v0.z; rB[h][3] = (__bf16)v0.w;
            rB[h][4] = (__bf16)v1.x; rB[h][5] = (__bf16)v1.y;
            rB[h][6] = (__bf16)v1.z; rB[h][7] = (__bf16)v1.w;
        }
    };
    auto storeTile = [&](int p) {
        *(bf16x8*)&As[p][lr][lc]     = rA[0];
        *(bf16x8*)&As[p][lr][lc + 8] = rA[1];
        *(bf16x8*)&Bs[p][lr][lc]     = rB[0];
        *(bf16x8*)&Bs[p][lr][lc + 8] = rB[1];
    };

    loadA(0); loadB(0); storeTile(0);
    __syncthreads();

    int p = 0;
    for (int k0 = 0; k0 < 1024; k0 += 32) {
        const bool more = (k0 + 32) < 1024;
        if (more) { loadA(k0 + 32); loadB(k0 + 32); }

        bf16x16 af[4], bfr[2];
#pragma unroll
        for (int mt = 0; mt < 4; ++mt)
            af[mt] = ldsA(&As[p][0][0], GP, 64 * wm + 16 * mt, 0, lane);
#pragma unroll
        for (int nt = 0; nt < 2; ++nt)
            bfr[nt] = ldsB(&Bs[p][0][0], GP, 32 * wn + 16 * nt, 0, lane);
#pragma unroll
        for (int mt = 0; mt < 4; ++mt)
#pragma unroll
            for (int nt = 0; nt < 2; ++nt)
                acc[mt][nt] = wmma_bf16(af[mt], bfr[nt], acc[mt][nt]);

        if (more) storeTile(p ^ 1);
        __syncthreads();
        p ^= 1;
    }

    // ---- epilogue: bias, activation, scatter ----
    const int hi8 = (lane >> 4) << 3;
    const int ln  = lane & 15;
#pragma unroll
    for (int mt = 0; mt < 4; ++mt) {
#pragma unroll
        for (int nt = 0; nt < 2; ++nt) {
            const int e = n0 + 32 * wn + 16 * nt + ln;
            const float bv = bias[e];
#pragma unroll
            for (int r = 0; r < 8; ++r) {
                const int m = m0 + 64 * wm + 16 * mt + r + hi8;
                float v = acc[mt][nt][r] + bv;
                if (act) v = feature_map(v);
                if (outHeadScatterBf16) {
                    const int n = m & 1, l = m >> 1;
                    ((__bf16*)outp)[((size_t)((n * 16 + (e >> 6)) * 1024 + l)) * 64 + (e & 63)]
                        = (__bf16)v;
                } else {
                    ((float*)outp)[(size_t)m * 1024 + e] = v;
                }
            }
        }
    }
}

// ---------------------------------------------------------------------------
// Chunked causal linear attention, one WG per (batch*head), 128 thr = 4 waves.
// Q/K chunk tiles are TDM-prefetched (double-buffered, DMA overlapped with
// the four chunk GEMMs); V loaded+transposed by VALU; state S fp32 in LDS.
// ---------------------------------------------------------------------------
#define PC 72                       // bf16 pitch (144 B rows, conflict-free)
#define QK_TILE (64 * PC)           // one Q or K buffer, elements
#define STP 65                      // fp32 pitch for state (260 B rows)

#define SM_QB   0
#define SM_KB   (SM_QB + 2 * QK_TILE * 2)          // bytes
#define SM_VT   (SM_KB + 2 * QK_TILE * 2)
#define SM_SC   (SM_VT + QK_TILE * 2)
#define SM_STF  (SM_SC + QK_TILE * 2)
#define SM_Z    (SM_STF + 64 * STP * 4)
#define SM_KCOL (SM_Z + 256)
#define SM_DEN  (SM_KCOL + 256)
#define SM_TOTAL (SM_DEN + 256)

__global__ __launch_bounds__(128) void linattn_kernel(
    const __bf16* __restrict__ Q, const __bf16* __restrict__ K,
    const __bf16* __restrict__ V, __bf16* __restrict__ O)
{
    extern __shared__ char smem[];
    __bf16* QbP  = (__bf16*)(smem + SM_QB);    // [2][64][PC]
    __bf16* KbP  = (__bf16*)(smem + SM_KB);    // [2][64][PC]
    __bf16* VTb  = (__bf16*)(smem + SM_VT);    // [64][PC]  V^T [e][s]
    __bf16* Sc   = (__bf16*)(smem + SM_SC);    // [64][PC]  masked scores [t][s]
    float*  STf  = (float*)(smem + SM_STF);    // [64][STP] S^T fp32 [e][d]
    float*  zS   = (float*)(smem + SM_Z);
    float*  kcol = (float*)(smem + SM_KCOL);
    float*  denom= (float*)(smem + SM_DEN);

    const int tid  = threadIdx.x;
    const int lane = tid & 31;
    const int w    = tid >> 5;          // wave 0..3
    const int hi8  = (lane >> 4) << 3;
    const int ln   = lane & 15;

    const size_t base = (size_t)blockIdx.x * 1024 * 64;
    const __bf16* Qg = Q + base;
    const __bf16* Kg = K + base;
    const __bf16* Vg = V + base;
    __bf16*       Og = O + base;

    const unsigned ldsQ[2] = { (unsigned)(uintptr_t)QbP,
                               (unsigned)(uintptr_t)(QbP + QK_TILE) };
    const unsigned ldsK[2] = { (unsigned)(uintptr_t)KbP,
                               (unsigned)(uintptr_t)(KbP + QK_TILE) };

    // init state
    for (int i = tid; i < 64 * STP; i += 128) STf[i] = 0.0f;
    if (tid < 64) zS[tid] = 0.0f;

    // kick off chunk 0 DMA (wave 0 only; uniform branch)
    if (w == 0) {
        tdm_load_2d_bf16_padded(Qg, ldsQ[0], 64, 64, 64);
        tdm_load_2d_bf16_padded(Kg, ldsK[0], 64, 64, 64);
    }
    __syncthreads();   // covers STf/z init; DMA still in flight

    for (int c = 0; c < 16; ++c) {
        const int t0 = c * 64;
        const int p  = c & 1;
        const __bf16* Qb = QbP + p * QK_TILE;
        const __bf16* Kb = KbP + p * QK_TILE;

        // ---- V chunk: load bf16, store transposed (prev GEMM4 done: barrier) ----
        {
            const int r = tid >> 1;
            const int h = (tid & 1) * 32;
            const __bf16* vp = Vg + (size_t)(t0 + r) * 64 + h;
#pragma unroll
            for (int j = 0; j < 32; j += 8) {
                bf16x8 v8 = *(const bf16x8*)(vp + j);
#pragma unroll
                for (int i = 0; i < 8; ++i) VTb[(h + j + i) * PC + r] = v8[i];
            }
        }

        // ---- prefetch next chunk via TDM, retire this chunk's DMA ----
        if (w == 0) {
            if (c < 15) {
                tdm_load_2d_bf16_padded(Qg + (size_t)(t0 + 64) * 64, ldsQ[p ^ 1], 64, 64, 64);
                tdm_load_2d_bf16_padded(Kg + (size_t)(t0 + 64) * 64, ldsK[p ^ 1], 64, 64, 64);
                __builtin_amdgcn_s_wait_tensorcnt(2);
            } else {
                __builtin_amdgcn_s_wait_tensorcnt(0);
            }
        }
        __syncthreads();   // Qb/Kb (buf p) + VTb ready for everyone

        // Q A-fragments for this wave's 16-row slab (reused in num GEMM)
        bf16x16 qa0 = ldsA(Qb, PC, 16 * w, 0, lane);
        bf16x16 qa1 = ldsA(Qb, PC, 16 * w, 32, lane);

        // ---- GEMM1: scores = Q K^T, causal mask, store bf16 ----
#pragma unroll
        for (int nt = 0; nt < 4; ++nt) {
            f32x8 a = zero8();
            a = wmma_bf16(qa0, ldsB(Kb, PC, 16 * nt, 0,  lane), a);
            a = wmma_bf16(qa1, ldsB(Kb, PC, 16 * nt, 32, lane), a);
            const int s  = 16 * nt + ln;
            const int tb = 16 * w + hi8;
#pragma unroll
            for (int r = 0; r < 8; ++r) {
                const int t = tb + r;
                Sc[t * PC + s] = (__bf16)((s <= t) ? a[r] : 0.0f);
            }
        }
        __syncthreads();

        // ---- denom (waves 0-1) and K column sums (waves 2-3) ----
        if (tid < 64) {
            float rs = 0.0f, qz = 0.0f;
#pragma unroll 8
            for (int s = 0; s < 64; ++s) rs += (float)Sc[tid * PC + s];
#pragma unroll 8
            for (int d = 0; d < 64; ++d) qz += (float)Qb[tid * PC + d] * zS[d];
            denom[tid] = rs + qz + 1e-6f;
        } else {
            const int d = tid - 64;
            float cs = 0.0f;
#pragma unroll 8
            for (int s = 0; s < 64; ++s) cs += (float)Kb[s * PC + d];
            kcol[d] = cs;
        }
        __syncthreads();
        if (tid < 64) zS[tid] += kcol[tid];   // wave-uniform branch

        // ---- GEMM2+3: num = scores V + Q S_prev ; divide ; store bf16 ----
        bf16x16 sa0 = ldsA(Sc, PC, 16 * w, 0, lane);
        bf16x16 sa1 = ldsA(Sc, PC, 16 * w, 32, lane);
#pragma unroll
        for (int nt = 0; nt < 4; ++nt) {
            f32x8 a = zero8();
            a = wmma_bf16(sa0, ldsB(VTb, PC, 16 * nt, 0,  lane), a);
            a = wmma_bf16(sa1, ldsB(VTb, PC, 16 * nt, 32, lane), a);
            a = wmma_bf16(qa0, ldsB_f32(STf, STP, 16 * nt, 0,  lane), a);
            a = wmma_bf16(qa1, ldsB_f32(STf, STP, 16 * nt, 32, lane), a);
            const int e  = 16 * nt + ln;
            const int tb = 16 * w + hi8;
#pragma unroll
            for (int r = 0; r < 8; ++r) {
                const int t = tb + r;
                Og[(size_t)(t0 + t) * 64 + e] = (__bf16)(a[r] / denom[t]);
            }
        }
        __syncthreads();   // all reads of STf/Sc done

        // ---- GEMM4: S += K^T V (fp32 accumulate in LDS) ----
        bf16x16 ka0, ka1;   // A = K^T : gather columns of Kb
        {
            const int dd = 16 * w + ln;
#pragma unroll
            for (int e = 0; e < 16; ++e) {
                const int s = ((e < 8) ? 0 : 16) + hi8 + (e & 7);
                ka0[e] = Kb[s * PC + dd];
                ka1[e] = Kb[(s + 32) * PC + dd];
            }
        }
#pragma unroll
        for (int nt = 0; nt < 4; ++nt) {
            f32x8 a = zero8();
            a = wmma_bf16(ka0, ldsB(VTb, PC, 16 * nt, 0,  lane), a);
            a = wmma_bf16(ka1, ldsB(VTb, PC, 16 * nt, 32, lane), a);
            const int e  = 16 * nt + ln;
            const int db = 16 * w + hi8;
#pragma unroll
            for (int r = 0; r < 8; ++r) {
                const int dd = db + r;
                STf[e * STP + dd] += a[r];
            }
        }
        __syncthreads();
    }
}

// ---------------------------------------------------------------------------
// launch
// ---------------------------------------------------------------------------
extern "C" void kernel_launch(void* const* d_in, const int* in_sizes, int n_in,
                              void* d_out, int out_size, void* d_ws, size_t ws_size,
                              hipStream_t stream) {
    const float* query = (const float*)d_in[0];
    const float* key_  = (const float*)d_in[1];
    const float* value = (const float*)d_in[2];
    const float* Wq = (const float*)d_in[3];
    const float* bq = (const float*)d_in[4];
    const float* Wk = (const float*)d_in[5];
    const float* bk = (const float*)d_in[6];
    const float* Wv = (const float*)d_in[7];
    const float* bv = (const float*)d_in[8];
    const float* Wo = (const float*)d_in[9];
    const float* bo = (const float*)d_in[10];
    float* out = (float*)d_out;

    __bf16* ws  = (__bf16*)d_ws;
    const size_t SZ = (size_t)2048 * 1024;   // one (L*N, E) activation, bf16
    __bf16* wsQ = ws;
    __bf16* wsK = ws + SZ;
    __bf16* wsV = ws + 2 * SZ;
    __bf16* wsA = ws + 3 * SZ;

    dim3 gGemm(8, 16);   // N/128 x M/128
    dim3 bGemm(256);
    // Q/K projections with phi(), V plain; all scattered bf16 to (B*H, L, d)
    proj_gemm_kernel<<<gGemm, bGemm, 0, stream>>>(query, Wq, bq, wsQ, 0, 1, 1);
    proj_gemm_kernel<<<gGemm, bGemm, 0, stream>>>(key_, Wk, bk, wsK, 0, 1, 1);
    proj_gemm_kernel<<<gGemm, bGemm, 0, stream>>>(value, Wv, bv, wsV, 0, 0, 1);

    // causal linear attention, one WG per head (dynamic LDS: 72.7 KB)
    linattn_kernel<<<dim3(32), dim3(128), SM_TOTAL, stream>>>(wsQ, wsK, wsV, wsA);

    // output projection: gather bf16 heads, write fp32 (L, N, E)
    proj_gemm_kernel<<<gGemm, bGemm, 0, stream>>>(wsA, Wo, bo, out, 1, 0, 0);
}
